// FlashMHA_31482110280462
// MI455X (gfx1250) — matrix operand order repaired and checked by reading, not compiled, not run on hardware
//
#include <hip/hip_runtime.h>
#include <hip/hip_bf16.h>

typedef __attribute__((ext_vector_type(16))) __bf16 v16bf;
typedef __attribute__((ext_vector_type(8)))  __bf16 v8bf;
typedef __attribute__((ext_vector_type(4)))  __bf16 v4bf;
typedef __attribute__((ext_vector_type(8)))  float  v8f;
typedef __attribute__((ext_vector_type(4)))  float  v4f;

#define EMBED 1024
#define NH    16
#define HD    64
#define BB    2
#define SQ    2048
#define SK    2048
#define KDIM  1024

static __device__ inline v16bf mk16(v8bf lo, v8bf hi) {
    return __builtin_shufflevector(lo, hi, 0,1,2,3,4,5,6,7,8,9,10,11,12,13,14,15);
}

static __device__ inline v8f wmma_bf16(v16bf a, v16bf b, v8f c) {
    // D = A(16x32 bf16) * B(32x16 bf16) + C(16x16 f32)
    return __builtin_amdgcn_wmma_f32_16x16x32_bf16(false, a, false, b, (short)0, c, false, false);
}

// --- CDNA5 async global->LDS copy (ASYNCcnt-tracked), per CDNA5 ISA ---
static __device__ inline void async_b128(void* lds_dst, const void* gsrc) {
    const unsigned int loff = (unsigned int)(size_t)lds_dst;   // low 32b of generic ptr = LDS offset
    asm volatile("global_load_async_to_lds_b128 %0, %1, off"
                 :: "v"(loff), "v"(gsrc) : "memory");
}
static __device__ inline void wait_async_le8() {
    asm volatile("s_wait_asynccnt 8" ::: "memory");
}
static __device__ inline void wait_async0() {
    asm volatile("s_wait_asynccnt 0" ::: "memory");
}

// ---------------------------------------------------------------------------
// Projection GEMM: C[M=4096][N] = A[M][1024] * W[N][1024]^T + bias[N]
// MODE 0: Q proj  -> (acc+b)*0.125 -> bf16 Qs[b][h][sq][64]
// MODE 1: KV proj -> n<1024: K[b][h][sk][64];  n>=1024: V^T[b][h][d][SK]
// MODE 2: Out proj (A is bf16 ctx) -> fp32 out[m][n]
// ---------------------------------------------------------------------------
template <int MODE>
__global__ __launch_bounds__(128)
void proj_kernel(const float* __restrict__ Af32,
                 const __bf16* __restrict__ Abf16,
                 const float* __restrict__ W,
                 const float* __restrict__ bias,
                 __bf16* __restrict__ outK,
                 __bf16* __restrict__ outV,
                 float*  __restrict__ outF)
{
    __shared__ __align__(16) __bf16 Alds[64 * 32];
    __shared__ __align__(16) __bf16 Wlds[64 * 32];

    const int tid  = threadIdx.x;
    const int w    = tid >> 5;
    const int lane = tid & 31;
    const int l15  = lane & 15;
    const int hi   = lane >> 4;
    const int row0 = blockIdx.y * 64;
    const int col0 = blockIdx.x * 64;

    v8f acc[4] = {};

    for (int kk = 0; kk < KDIM; kk += 32) {
        __syncthreads();
        for (int i = tid; i < 512; i += 128) {          // 64 rows x 8 x (4 elems)
            const int r = i >> 3, c4 = (i & 7) * 4;
            v4bf ab;
            if (MODE == 2) {
                ab = *(const v4bf*)(Abf16 + (size_t)(row0 + r) * KDIM + kk + c4);
            } else {
                const v4f av = *(const v4f*)(Af32 + (size_t)(row0 + r) * KDIM + kk + c4);
#pragma unroll
                for (int j = 0; j < 4; ++j) ab[j] = (__bf16)av[j];
            }
            *(v4bf*)(&Alds[r * 32 + c4]) = ab;
            const v4f wv = *(const v4f*)(W + (size_t)(col0 + r) * KDIM + kk + c4);
            v4bf wb;
#pragma unroll
            for (int j = 0; j < 4; ++j) wb[j] = (__bf16)wv[j];
            *(v4bf*)(&Wlds[r * 32 + c4]) = wb;
        }
        __syncthreads();

        const __bf16* ar = &Alds[(w * 16 + l15) * 32];
        const v16bf a = mk16(*(const v8bf*)(ar + hi * 16),
                             *(const v8bf*)(ar + hi * 16 + 8));
#pragma unroll
        for (int nt = 0; nt < 4; ++nt) {
            const __bf16* br = &Wlds[(nt * 16 + l15) * 32 + hi * 16];
            const v16bf bf = mk16(*(const v8bf*)br, *(const v8bf*)(br + 8));
            acc[nt] = wmma_bf16(a, bf, acc[nt]);
        }
    }

#pragma unroll
    for (int nt = 0; nt < 4; ++nt) {
        const int n  = col0 + nt * 16 + l15;
        const float bv = bias[n];
#pragma unroll
        for (int r = 0; r < 8; ++r) {
            const int m = row0 + w * 16 + r + 8 * hi;   // global row (b*2048 + s)
            float v = acc[nt][r] + bv;
            if (MODE == 0) {
                v *= 0.125f;                            // 1/sqrt(HD) folded into Q
                const int b = m >> 11, sq = m & 2047;
                const int h = n >> 6,  d  = n & 63;
                outK[((((size_t)b * NH + h) * SQ) + sq) * HD + d] = (__bf16)v;
            } else if (MODE == 1) {
                const int b = m >> 11, sk = m & 2047;
                if (n < 1024) {
                    const int h = n >> 6, d = n & 63;
                    outK[((((size_t)b * NH + h) * SK) + sk) * HD + d] = (__bf16)v;
                } else {
                    const int n2 = n - 1024;
                    const int h = n2 >> 6, d = n2 & 63;
                    outV[((((size_t)b * NH + h) * HD) + d) * SK + sk] = (__bf16)v;  // V^T
                }
            } else {
                outF[(size_t)m * EMBED + n] = v;
            }
        }
    }
}

// ---------------------------------------------------------------------------
// Flash attention: 128 threads = 4 waves; WG covers (b,h) x 64 sq rows, each
// wave one 16-row tile. Streams SK in 64-wide chunks with double-buffered
// async global->LDS K/V tiles; row-sum via WMMA against an all-ones fragment.
// ---------------------------------------------------------------------------
__global__ __launch_bounds__(128)
void attn_kernel(const __bf16* __restrict__ Qs,
                 const __bf16* __restrict__ Kb,
                 const __bf16* __restrict__ Vt,
                 const float* __restrict__ attn_bias,
                 const int* __restrict__ mask,
                 __bf16* __restrict__ Ctx)
{
    __shared__ __align__(16) __bf16 Qlds[64 * 64];
    __shared__ __align__(16) __bf16 Klds[2][64 * 64];
    __shared__ __align__(16) __bf16 Vlds[2][64 * 64];
    __shared__ __align__(16) __bf16 Plds[4][16 * 64];
    __shared__ float CB[64];

    const int tid  = threadIdx.x;
    const int w    = tid >> 5;
    const int lane = tid & 31;
    const int l15  = lane & 15;
    const int hi   = lane >> 4;
    const int bh   = blockIdx.y;
    const int b    = bh >> 4;
    const int sqB  = blockIdx.x * 64;

    const __bf16* Kgb = Kb + (size_t)bh * SK * HD;
    const __bf16* Vgb = Vt + (size_t)bh * HD * SK;

    // ---- async preload: Q tile (4 ops/thread), then K/V tile 0 (8 ops/thread)
    {
        const __bf16* Qg = Qs + ((size_t)bh * SQ + sqB) * HD;
        for (int i = tid; i < 512; i += 128)
            async_b128(&Qlds[i * 8], Qg + i * 8);
    }
    {
        for (int i = tid; i < 512; i += 128)                   // K rows contiguous
            async_b128(&Klds[0][i * 8], Kgb + (size_t)0 * HD + i * 8);
        for (int i = tid; i < 512; i += 128) {
            const int d = i >> 3, c = (i & 7) * 8;
            async_b128(&Vlds[0][d * 64 + c], Vgb + (size_t)d * SK + 0 + c);
        }
    }
    wait_async0();
    __syncthreads();

    v16bf qa[2];
    {
        const __bf16* qr = &Qlds[(w * 16 + l15) * 64];
#pragma unroll
        for (int kt = 0; kt < 2; ++kt)
            qa[kt] = mk16(*(const v8bf*)(qr + kt * 32 + hi * 16),
                          *(const v8bf*)(qr + kt * 32 + hi * 16 + 8));
    }
    v16bf onesf;
#pragma unroll
    for (int i = 0; i < 16; ++i) onesf[i] = (__bf16)1.0f;

    v8f ctx[4] = {};
    v8f lsum = {};
    float mrow[8];
#pragma unroll
    for (int r = 0; r < 8; ++r) mrow[r] = -1e30f;

    for (int sk0 = 0; sk0 < SK; sk0 += 64) {
        const int cur = (sk0 >> 6) & 1;
        __syncthreads();   // A: all reads of buffer cur^1 (prev iter) are done

        if (sk0 + 64 < SK) {                    // issue next tile into cur^1
            const int nx = sk0 + 64;
            for (int i = tid; i < 512; i += 128)
                async_b128(&Klds[cur ^ 1][i * 8], Kgb + (size_t)nx * HD + i * 8);
            for (int i = tid; i < 512; i += 128) {
                const int d = i >> 3, c = (i & 7) * 8;
                async_b128(&Vlds[cur ^ 1][d * 64 + c], Vgb + (size_t)d * SK + nx + c);
            }
            wait_async_le8();                   // 8 newest = next tile; tile cur done
        } else {
            wait_async0();
        }
        if (tid < 64) {                         // combined bias+mask for this chunk
            const int j = sk0 + tid;
            CB[tid] = attn_bias[(size_t)b * SK + j] +
                      (mask[(size_t)b * SK + j] ? 0.0f : -1e30f);
        }
        __syncthreads();   // B: tile cur resident in LDS for all threads

        // S = Q K^T : 4 N-tiles x 2 K-steps
        v8f sfr[4];
#pragma unroll
        for (int nt = 0; nt < 4; ++nt) {
            v8f s = {};
#pragma unroll
            for (int kt = 0; kt < 2; ++kt) {
                const __bf16* kr = &Klds[cur][(nt * 16 + l15) * 64 + kt * 32 + hi * 16];
                s = wmma_bf16(qa[kt], mk16(*(const v8bf*)kr, *(const v8bf*)(kr + 8)), s);
            }
            const float bv = CB[nt * 16 + l15];
#pragma unroll
            for (int r = 0; r < 8; ++r) s[r] += bv;
            sfr[nt] = s;
        }

        // online softmax: row max over 64 cols (4 tiles + 16-lane butterfly)
#pragma unroll
        for (int r = 0; r < 8; ++r) {
            float cm = fmaxf(fmaxf(sfr[0][r], sfr[1][r]), fmaxf(sfr[2][r], sfr[3][r]));
            for (int off = 1; off < 16; off <<= 1) cm = fmaxf(cm, __shfl_xor(cm, off, 16));
            const float mn = fmaxf(mrow[r], cm);
            const float sc = __expf(mrow[r] - mn);
            mrow[r] = mn;
            lsum[r] *= sc;
#pragma unroll
            for (int dt = 0; dt < 4; ++dt) ctx[dt][r] *= sc;
        }
#pragma unroll
        for (int nt = 0; nt < 4; ++nt)
#pragma unroll
            for (int r = 0; r < 8; ++r) sfr[nt][r] = __expf(sfr[nt][r] - mrow[r]);

        // re-stripe P: C-layout -> LDS (16x64 row-major) -> A-layout
#pragma unroll
        for (int nt = 0; nt < 4; ++nt)
#pragma unroll
            for (int r = 0; r < 8; ++r)
                Plds[w][(r + 8 * hi) * 64 + nt * 16 + l15] = (__bf16)sfr[nt][r];
        __syncthreads();   // C: P visible wave-wide

        const __bf16* pr = &Plds[w][l15 * 64];
        v16bf pa[2];
#pragma unroll
        for (int kt = 0; kt < 2; ++kt)
            pa[kt] = mk16(*(const v8bf*)(pr + kt * 32 + hi * 16),
                          *(const v8bf*)(pr + kt * 32 + hi * 16 + 8));
#pragma unroll
        for (int dt = 0; dt < 4; ++dt)
#pragma unroll
            for (int kt = 0; kt < 2; ++kt) {
                const __bf16* vr = &Vlds[cur][(dt * 16 + l15) * 64 + kt * 32 + hi * 16];
                ctx[dt] = wmma_bf16(pa[kt],
                                    mk16(*(const v8bf*)vr, *(const v8bf*)(vr + 8)),
                                    ctx[dt]);
            }
        // row-sum via matrix pipe: l += P * ones  (lands in C-frag row layout)
        lsum = wmma_bf16(pa[0], onesf, lsum);
        lsum = wmma_bf16(pa[1], onesf, lsum);
    }

    // normalize and store ctx as bf16 [b][sq][h*64+d]
#pragma unroll
    for (int dt = 0; dt < 4; ++dt)
#pragma unroll
        for (int r = 0; r < 8; ++r) {
            const float v = ctx[dt][r] / lsum[r];
            const int sq = sqB + w * 16 + r + 8 * hi;
            const int e  = (bh & 15) * HD + dt * 16 + l15;
            Ctx[((size_t)b * SQ + sq) * EMBED + e] = (__bf16)v;
        }
}

extern "C" void kernel_launch(void* const* d_in, const int* in_sizes, int n_in,
                              void* d_out, int out_size, void* d_ws, size_t ws_size,
                              hipStream_t stream) {
    const float* tgt   = (const float*)d_in[0];
    const float* mem   = (const float*)d_in[1];
    const float* abias = (const float*)d_in[2];
    const int*   mask  = (const int*)d_in[3];
    const float* Wq_w  = (const float*)d_in[4];
    const float* Wq_b  = (const float*)d_in[5];
    const float* Wkv_w = (const float*)d_in[6];
    const float* Wkv_b = (const float*)d_in[7];
    const float* out_w = (const float*)d_in[8];
    const float* out_b = (const float*)d_in[9];
    float* out = (float*)d_out;

    char* ws = (char*)d_ws;
    __bf16* Qs  = (__bf16*)(ws);                         // 8 MiB: [B][H][SQ][64]
    __bf16* Kb  = (__bf16*)(ws + (size_t)(8  << 20));    // 8 MiB: [B][H][SK][64]
    __bf16* Vt  = (__bf16*)(ws + (size_t)(16 << 20));    // 8 MiB: [B][H][64][SK]
    __bf16* Ctx = (__bf16*)(ws + (size_t)(24 << 20));    // 8 MiB: [B][SQ][1024]

    const dim3 blk(128);
    // Q projection: M=4096, N=1024
    proj_kernel<0><<<dim3(16, 64), blk, 0, stream>>>(tgt, nullptr, Wq_w, Wq_b, Qs, nullptr, nullptr);
    // KV projection: M=4096, N=2048
    proj_kernel<1><<<dim3(32, 64), blk, 0, stream>>>(mem, nullptr, Wkv_w, Wkv_b, Kb, Vt, nullptr);
    // Flash attention: grid (SQ/64, B*H)
    attn_kernel<<<dim3(SQ / 64, BB * NH), blk, 0, stream>>>(Qs, Kb, Vt, abias, mask, Ctx);
    // Output projection: M=4096, N=1024, fp32 out
    proj_kernel<2><<<dim3(16, 64), blk, 0, stream>>>(nullptr, Ctx, out_w, out_b, nullptr, nullptr, out);
}
